// Cross_Attention_15453292331532
// MI455X (gfx1250) — matrix operand order repaired and checked by reading, not compiled, not run on hardware
//
#include <hip/hip_runtime.h>
#include <hip/hip_bf16.h>

// ---------------------------------------------------------------------------
// Cross attention, MI455X (gfx1250, wave32, WMMA bf16 16x16x32, f32 accum)
//   q = Wq@xt+bq  -> qT [B][N][64] bf16   (transposed for contiguous WMMA ops)
//   k = Wk@xs+bk  -> kT [B][N][64] bf16
//   v = Wv@xs+bv  -> v  [B][512][N] bf16  (j-contiguous for PV B-operand)
//   out = gamma * (V @ softmax(qT k)^T) + x_s   (flash-style, attn never hits HBM)
// ---------------------------------------------------------------------------

typedef __attribute__((ext_vector_type(16))) __bf16 v16bf;
typedef __attribute__((ext_vector_type(8)))  float  v8f;

#define NQ  4096   // W*H
#define CCH 512    // channels
#define DK  64     // q/k head dim

__device__ __forceinline__ v8f wmma_bf16(v16bf a, v16bf b, v8f c) {
  // 8 args: (neg_a, A, neg_b, B, c_mod, C, reuse_a, reuse_b)
  return __builtin_amdgcn_wmma_f32_16x16x32_bf16(false, a, false, b,
                                                 (short)0, c, false, false);
}

// ---------------------------------------------------------------------------
// Projection GEMM: out[b,m,n] = sum_c W[m,c] * x[b,c,n] + bias[m], bf16 out.
// One wave computes a 16(m) x 64(n) tile: 4 accumulators share one A operand
// per K-chunk (amortizes fp32 W loads + bf16 converts 4x).
// TRANSPOSE: 0 -> out[b][m*N+n] ; 1 -> out[b][n*M+m]
// ---------------------------------------------------------------------------
template <int TRANSPOSE>
__global__ __launch_bounds__(32)
void proj_kernel(const float* __restrict__ W, const float* __restrict__ bias,
                 const float* __restrict__ x, __bf16* __restrict__ out,
                 int M, int C, int N)
{
  const int lane  = threadIdx.x & 31;
  const int row16 = lane & 15;      // A: matrix row / B: matrix col
  const int half  = lane >> 4;      // lane group selects K sub-range
  const int n0 = blockIdx.x * 64;
  const int m0 = blockIdx.y * 16;
  const int b  = blockIdx.z;
  const float* xb = x + (size_t)b * C * N;

  v8f zero = {};
  v8f acc[4];
#pragma unroll
  for (int nt = 0; nt < 4; ++nt) acc[nt] = zero;

  for (int kc = 0; kc < C; kc += 32) {
    // A operand (16x32 bf16): lane row m=row16; K = half*16 + e
    v16bf a;
    const float* wrow = W + (size_t)(m0 + row16) * C + kc + half * 16;
#pragma unroll
    for (int e = 0; e < 8; ++e) {
      a[e]     = (__bf16)wrow[e];
      a[e + 8] = (__bf16)wrow[8 + e];
    }
#pragma unroll
    for (int nt = 0; nt < 4; ++nt) {
      // B operand (32x16 bf16): lane col n; K = half*16 + e (strided in x,
      // coalesced across lanes: 16 consecutive n per half-wave)
      v16bf bm;
      const float* xcol = xb + (size_t)(kc + half * 16) * N + (n0 + nt * 16 + row16);
#pragma unroll
      for (int e = 0; e < 16; ++e) bm[e] = (__bf16)xcol[(size_t)e * N];
      acc[nt] = wmma_bf16(a, bm, acc[nt]);
    }
  }

  // D layout: reg r -> row m0 + r + half*8 ; col = tile base + row16
  __bf16* outb = out + (size_t)b * M * N;
  const float bcol = 0.f;
#pragma unroll
  for (int nt = 0; nt < 4; ++nt) {
    const int n = n0 + nt * 16 + row16;
#pragma unroll
    for (int r = 0; r < 8; ++r) {
      const int m = m0 + r + half * 8;
      const float val = acc[nt][r] + bias[m] + bcol;
      if (TRANSPOSE) outb[(size_t)n * M + m] = (__bf16)val;
      else           outb[(size_t)m * N + n] = (__bf16)val;
    }
  }
}

// ---------------------------------------------------------------------------
// Flash attention + epilogue. One block = 8 wave32 = one (batch, 16-query) tile.
// Phase 1: online softmax stats (key tiles split across waves, merged in LDS).
// Phase 2: each wave produces one P tile (bf16, LDS) per super-iter, then all
//          waves consume 8 slots as 4 K=32 chunks against their 64 channels.
// v (4 MB/batch) is shared by all 256 blocks of a batch through the 192MB L2.
// ---------------------------------------------------------------------------
__global__ __launch_bounds__(256)
void attn_kernel(const __bf16* __restrict__ qT,   // [B][N][64]
                 const __bf16* __restrict__ kT,   // [B][N][64]
                 const __bf16* __restrict__ v,    // [B][512][N]
                 const float*  __restrict__ x_s,  // [B][512][N]
                 const float*  __restrict__ gamma,
                 float* __restrict__ out)         // [B][512][N]
{
  __shared__ __bf16 Pslots[8][16][16];   // P tiles, row=query, col=key
  __shared__ float  ms[8][16], ls[8][16];
  __shared__ float  mfin[16], lfin[16];

  const int tid   = threadIdx.x;
  const int wave  = tid >> 5;
  const int lane  = tid & 31;
  const int row16 = lane & 15;
  const int half  = lane >> 4;
  const int b  = blockIdx.y;
  const int i0 = blockIdx.x * 16;        // query tile base

  const __bf16* qTb = qT + (size_t)b * NQ * DK;
  const __bf16* kTb = kT + (size_t)b * NQ * DK;
  const __bf16* vb  = v  + (size_t)b * CCH * NQ;

  // --- q A-operands for K-chunks d=0..31 and d=32..63 (resident all kernel) ---
  v16bf aq0, aq1;
  {
    const __bf16* qrow = qTb + (size_t)(i0 + row16) * DK + half * 16;
#pragma unroll
    for (int e = 0; e < 8; ++e) {
      aq0[e]     = qrow[e];        aq0[e + 8] = qrow[8 + e];
      aq1[e]     = qrow[32 + e];   aq1[e + 8] = qrow[40 + e];
    }
  }

  // ---------------- Phase 1: row max / row sum over all 4096 keys ----------
  float mrun[8], lrun[8];
#pragma unroll
  for (int r = 0; r < 8; ++r) { mrun[r] = -1e30f; lrun[r] = 0.f; }

  for (int jt = wave; jt < NQ / 16; jt += 8) {
    const int j0 = jt * 16;
    const __bf16* krow = kTb + (size_t)(j0 + row16) * DK + half * 16;
    __builtin_prefetch(krow + 8 * 16 * DK, 0, 0);   // next tile for this wave
    v16bf bk0 = *(const v16bf*)krow;
    v16bf bk1 = *(const v16bf*)(krow + 32);
    v8f s = {};
    s = wmma_bf16(aq0, bk0, s);
    s = wmma_bf16(aq1, bk1, s);
#pragma unroll
    for (int r = 0; r < 8; ++r) {
      float xv = s[r];
      float mx = xv;
      for (int off = 1; off < 16; off <<= 1) mx = fmaxf(mx, __shfl_xor(mx, off));
      float mnew = fmaxf(mrun[r], mx);
      float p = __expf(xv - mnew);
      float ps = p;
      for (int off = 1; off < 16; off <<= 1) ps += __shfl_xor(ps, off);
      lrun[r] = lrun[r] * __expf(mrun[r] - mnew) + ps;
      mrun[r] = mnew;
    }
  }
  if (row16 == 0) {  // lanes 0 and 16 carry rows r and r+8
#pragma unroll
    for (int r = 0; r < 8; ++r) {
      ms[wave][r + half * 8] = mrun[r];
      ls[wave][r + half * 8] = lrun[r];
    }
  }
  __syncthreads();
  if (tid < 16) {
    float m = -1e30f;
    for (int w2 = 0; w2 < 8; ++w2) m = fmaxf(m, ms[w2][tid]);
    float l = 0.f;
    for (int w2 = 0; w2 < 8; ++w2) l += ls[w2][tid] * __expf(ms[w2][tid] - m);
    mfin[tid] = m;
    lfin[tid] = l;
  }
  __syncthreads();

  // ---------------- Phase 2: P production + PV accumulation ----------------
  v8f zero = {};
  v8f acc[4];
#pragma unroll
  for (int t = 0; t < 4; ++t) acc[t] = zero;
  const int c0w = wave * 64;             // this wave's 64 channels
  float mrow[8];
#pragma unroll
  for (int r = 0; r < 8; ++r) mrow[r] = mfin[r + half * 8];

  for (int t = 0; t < NQ / (16 * 8); ++t) {
    // produce P tile for key tile (t*8 + wave) into slot `wave`
    {
      const int j0 = (t * 8 + wave) * 16;
      const __bf16* krow = kTb + (size_t)(j0 + row16) * DK + half * 16;
      v16bf bk0 = *(const v16bf*)krow;
      v16bf bk1 = *(const v16bf*)(krow + 32);
      v8f s = {};
      s = wmma_bf16(aq0, bk0, s);
      s = wmma_bf16(aq1, bk1, s);
#pragma unroll
      for (int r = 0; r < 8; ++r) {
        float p = __expf(s[r] - mrow[r]);
        Pslots[wave][r + half * 8][row16] = (__bf16)p;
      }
    }
    __syncthreads();

    // consume the 8 slots as 4 K=32 chunks
#pragma unroll
    for (int kc = 0; kc < 4; ++kc) {
      const int jbase = (t * 8 + kc * 2) * 16;
      v16bf ap;   // A operand: half h reads P slot 2kc+h
#pragma unroll
      for (int e = 0; e < 8; ++e) {
        ap[e]     = Pslots[kc * 2 + half][row16][e];
        ap[e + 8] = Pslots[kc * 2 + half][row16][8 + e];
      }
#pragma unroll
      for (int ct = 0; ct < 4; ++ct) {
        const int c0 = c0w + ct * 16;
        // B operand: lane col = channel c0+row16, K = j contiguous -> one 32B load
        const __bf16* vrow = vb + (size_t)(c0 + row16) * NQ + jbase + half * 16;
        v16bf bvv = *(const v16bf*)vrow;
        acc[ct] = wmma_bf16(ap, bvv, acc[ct]);
      }
    }
    __syncthreads();
  }

  // ---------------- Epilogue: out = gamma * O/l + x_s ----------------------
  const float g = gamma[0];
  float linv[8];
#pragma unroll
  for (int r = 0; r < 8; ++r) linv[r] = 1.0f / lfin[r + half * 8];
  const float* xsb  = x_s + (size_t)b * CCH * NQ;
  float*       outb = out + (size_t)b * CCH * NQ;
#pragma unroll
  for (int ct = 0; ct < 4; ++ct) {
    const int ch = c0w + ct * 16 + row16;     // D col = channel
#pragma unroll
    for (int r = 0; r < 8; ++r) {
      const int q = r + half * 8;             // D row = query
      const size_t idx = (size_t)ch * NQ + (i0 + q);
      const float o = acc[ct][r] * linv[r];
      outb[idx] = g * o + xsb[idx];
    }
  }
}

// ---------------------------------------------------------------------------
extern "C" void kernel_launch(void* const* d_in, const int* in_sizes, int n_in,
                              void* d_out, int out_size, void* d_ws, size_t ws_size,
                              hipStream_t stream) {
  const float* x_s   = (const float*)d_in[0];
  const float* x_t   = (const float*)d_in[1];
  const float* Wq    = (const float*)d_in[2];
  const float* bq    = (const float*)d_in[3];
  const float* Wk    = (const float*)d_in[4];
  const float* bk    = (const float*)d_in[5];
  const float* Wv    = (const float*)d_in[6];
  const float* bv    = (const float*)d_in[7];
  const float* gamma = (const float*)d_in[8];
  float* out = (float*)d_out;

  // workspace: qT (2MB) | kT (2MB) | v (16MB)  = 20MB bf16
  char* ws = (char*)d_ws;
  __bf16* qT = (__bf16*)(ws);
  __bf16* kT = (__bf16*)(ws + (size_t)4 * NQ * DK * 2);
  __bf16* vW = (__bf16*)(ws + (size_t)8 * NQ * DK * 2);

  // q = Wq @ x_t (+bq), stored transposed [B][N][64]
  proj_kernel<1><<<dim3(NQ / 64, DK / 16, 4), 32, 0, stream>>>(
      Wq, bq, x_t, qT, DK, CCH, NQ);
  // k = Wk @ x_s (+bk), stored transposed [B][N][64]
  proj_kernel<1><<<dim3(NQ / 64, DK / 16, 4), 32, 0, stream>>>(
      Wk, bk, x_s, kT, DK, CCH, NQ);
  // v = Wv @ x_s (+bv), stored [B][512][N]
  proj_kernel<0><<<dim3(NQ / 64, CCH / 16, 4), 32, 0, stream>>>(
      Wv, bv, x_s, vW, CCH, CCH, NQ);
  // flash attention + residual epilogue
  attn_kernel<<<dim3(NQ / 16, 4), 256, 0, stream>>>(
      qT, kT, vW, x_s, gamma, out);
}
